// PageAttention_41308995453592
// MI455X (gfx1250) — matrix operand order repaired and checked by reading, not compiled, not run on hardware
//
#include <hip/hip_runtime.h>
#include <hip/hip_bf16.h>

typedef unsigned short u16_t;
typedef __attribute__((ext_vector_type(16))) __bf16 v16bf;
typedef __attribute__((ext_vector_type(8)))  float  v8f;

#define NB 8
#define NS 32
#define NHID 2048
#define NHEAD 16
#define NKVH 2
#define HDIM 128
#define KVL 8192
#define NGRP 8
#define ATT_SCALE 0.08838834764831845f

struct B32x8 { uint4 lo; uint4 hi; };

static __device__ __forceinline__ u16_t f2bf(float f) {
  unsigned u = __builtin_bit_cast(unsigned, f);
  unsigned r = u + 0x7fffu + ((u >> 16) & 1u);
  return (u16_t)(r >> 16);
}

static __device__ __forceinline__ unsigned pack2(float a, float b) {
  return (unsigned)f2bf(a) | ((unsigned)f2bf(b) << 16);
}

// Build a 16-element bf16 fragment from two 16-byte-aligned 8-element runs.
static __device__ __forceinline__ v16bf ldfrag(const u16_t* p0, const u16_t* p1) {
  B32x8 t;
  t.lo = *(const uint4*)p0;
  t.hi = *(const uint4*)p1;
  return __builtin_bit_cast(v16bf, t);
}

static __device__ __forceinline__ v8f wmma_bf16(v16bf a, v16bf b, v8f c) {
  return __builtin_amdgcn_wmma_f32_16x16x32_bf16(false, a, false, b, (short)0, c,
                                                 false, false);
}

// Async copy 64B (per lane, 4 x b128) from global bf16 to LDS; INST_OFFSET applies to both.
static __device__ __forceinline__ void async_b128(unsigned lds_addr,
                                                  const u16_t* gaddr) {
  unsigned long long g = (unsigned long long)gaddr;
  asm volatile("global_load_async_to_lds_b128 %0, %1, off offset:0" ::"v"(lds_addr),
               "v"(g)
               : "memory");
  asm volatile("global_load_async_to_lds_b128 %0, %1, off offset:16" ::"v"(lds_addr),
               "v"(g)
               : "memory");
  asm volatile("global_load_async_to_lds_b128 %0, %1, off offset:32" ::"v"(lds_addr),
               "v"(g)
               : "memory");
  asm volatile("global_load_async_to_lds_b128 %0, %1, off offset:48" ::"v"(lds_addr),
               "v"(g)
               : "memory");
}

static __device__ __forceinline__ void wait_async0() {
  asm volatile("s_wait_asynccnt 0x0" ::: "memory");
}

// ---------------------------------------------------------------- cvt f32->bf16
__global__ void cvt_kernel(const float* __restrict__ src, u16_t* __restrict__ dst,
                           int n8) {  // n8 = n/8
  for (int i = blockIdx.x * blockDim.x + threadIdx.x; i < n8;
       i += gridDim.x * blockDim.x) {
    const float4* s = (const float4*)(src + (size_t)i * 8);
    float4 a = s[0], b = s[1];
    uint4 o;
    o.x = pack2(a.x, a.y);
    o.y = pack2(a.z, a.w);
    o.z = pack2(b.x, b.y);
    o.w = pack2(b.z, b.w);
    *(uint4*)(dst + (size_t)i * 8) = o;
  }
}

// -------------------------- K/V cache -> bf16 copy with fresh-row splice at rp
// cache layout (b, kvh, pos, d); fresh layout (b, kvh, s, d) already bf16.
__global__ void cvt_kv(const float* __restrict__ cache,
                       const u16_t* __restrict__ fresh, u16_t* __restrict__ dst,
                       const int* __restrict__ rp_ptr, int total8) {
  int t = blockIdx.x * blockDim.x + threadIdx.x;
  if (t >= total8) return;
  const size_t base = (size_t)t * 8;
  const int d = (int)(base & (HDIM - 1));
  const size_t row = base >> 7;          // (b,kvh,pos)
  const int pos = (int)(row & (KVL - 1));
  const size_t bk = row >> 13;           // b*NKVH + kvh
  const int rp = rp_ptr[0];
  uint4 o;
  if (pos >= rp && pos < rp + NS) {
    o = *(const uint4*)(fresh + ((bk * NS + (size_t)(pos - rp)) << 7) + d);
  } else {
    const float4* s = (const float4*)(cache + base);
    float4 a = s[0], b = s[1];
    o.x = pack2(a.x, a.y);
    o.y = pack2(a.z, a.w);
    o.z = pack2(b.x, b.y);
    o.w = pack2(b.z, b.w);
  }
  *(uint4*)(dst + base) = o;
}

// ------------------------------------------------- generic bf16 WMMA GEMM + bias
// C[M,N](f32) = A[M,K](bf16) * B[K,N](bf16) + bias[N]
__global__ void __launch_bounds__(128)
gemm_bf16(const u16_t* __restrict__ A, const u16_t* __restrict__ Bm,
          const float* __restrict__ bias, float* __restrict__ C,
          int M, int N, int K) {
  __shared__ u16_t As[64][40];
  __shared__ u16_t Bt[64][40];
  const int tid = threadIdx.x;
  const int w = tid >> 5, lane = tid & 31, l15 = lane & 15, hf = lane >> 4;
  const int M0 = blockIdx.x * 64, N0 = blockIdx.y * 64;

  v8f acc[4];
#pragma unroll
  for (int i = 0; i < 4; ++i) acc[i] = (v8f){0, 0, 0, 0, 0, 0, 0, 0};

  for (int k0 = 0; k0 < K; k0 += 32) {
    {
      const int row = tid >> 1, c0 = (tid & 1) * 16;
      const u16_t* s = A + (size_t)(M0 + row) * K + k0 + c0;
      *(uint4*)&As[row][c0] = ((const uint4*)s)[0];
      *(uint4*)&As[row][c0 + 8] = ((const uint4*)s)[1];
    }
    {
      const int krow = tid >> 2, n0 = (tid & 3) * 16;
      const u16_t* s = Bm + (size_t)(k0 + krow) * N + N0 + n0;
#pragma unroll
      for (int e = 0; e < 16; ++e) Bt[n0 + e][krow] = s[e];
    }
    __syncthreads();

    const u16_t* ar = &As[w * 16 + l15][0];
    v16bf af = ldfrag(ar + hf * 8, ar + 16 + hf * 8);
#pragma unroll
    for (int nf = 0; nf < 4; ++nf) {
      const u16_t* br = &Bt[nf * 16 + l15][hf * 8];
      v16bf bf_ = ldfrag(br, br + 16);
      acc[nf] = wmma_bf16(af, bf_, acc[nf]);
    }
    __syncthreads();
  }

#pragma unroll
  for (int nf = 0; nf < 4; ++nf)
#pragma unroll
    for (int r = 0; r < 8; ++r) {
      const int m = M0 + w * 16 + r + 8 * hf;
      const int n = N0 + nf * 16 + l15;
      float v = acc[nf][r] + (bias ? bias[n] : 0.0f);
      C[(size_t)m * N + n] = v;
    }
}

// ----------------------------------------------- RoPE + repack to (b,h,s,d) bf16
__global__ void rope_pack(const float* __restrict__ src,
                          const float* __restrict__ cosb,
                          const float* __restrict__ sinb,
                          u16_t* __restrict__ dst, int nheads, int mode,
                          int total) {
  int t = blockIdx.x * blockDim.x + threadIdx.x;
  if (t >= total) return;
  const int d = t & 63;
  int tmp = t >> 6;
  const int hh = tmp % nheads;
  tmp /= nheads;
  const int s = tmp % NS;
  const int b = tmp / NS;

  const float* row = src + (size_t)(b * NS + s) * (nheads * HDIM) + hh * HDIM;
  const float x1 = row[d], x2 = row[d + 64];
  float o1, o2;
  if (mode == 1) {
    o1 = x1;
    o2 = x2;
  } else {
    const size_t cb = (size_t)(b * NS + s) * HDIM;
    const float c1 = cosb[cb + d], s1 = sinb[cb + d];
    const float c2 = cosb[cb + d + 64], s2 = sinb[cb + d + 64];
    o1 = x1 * c1 - x2 * s1;
    o2 = x2 * c2 + x1 * s2;
  }
  u16_t* drow = dst + (((size_t)b * nheads + hh) * NS + s) * HDIM;
  drow[d] = f2bf(o1);
  drow[d + 64] = f2bf(o2);
}

// --------------------------------------------------- flash attention per (b,head)
// 8 waves / 256 threads, KV chunk = 64. Waves w=0,4 (dt==0) run QK^T + online
// softmax; all waves run P*V for a 32-wide HDIM slice. K chunk arrives via
// global_load_async_to_lds_b128 (bf16, pre-converted), V is staged transposed.
__global__ void __launch_bounds__(256)
attn_kernel(const u16_t* __restrict__ qbf, const u16_t* __restrict__ kbf,
            const u16_t* __restrict__ vbf, const float* __restrict__ mask,
            u16_t* __restrict__ abf) {
  __shared__ u16_t Ks[64][136];   // kv-row major K chunk, +16B pad
  __shared__ u16_t Vt[128][72];   // d-major (transposed) V chunk, +16B pad
  __shared__ u16_t Ps[32][72];    // probabilities (bf16)
  __shared__ float cfac[32];
  __shared__ float lsum[32];

  const int bh = blockIdx.x;
  const int b = bh / NHEAD, hh = bh % NHEAD, kh = hh / NGRP;
  const int tid = threadIdx.x;
  const int w = tid >> 5, lane = tid & 31, l15 = lane & 15, hf = lane >> 4;
  const int qt = w >> 2, dt = w & 3;

  v16bf Aq[4];
  float m_run[8], l_run[8];
  if (dt == 0) {
    const u16_t* qr = qbf + (((size_t)b * NHEAD + hh) * NS + qt * 16 + l15) * HDIM;
#pragma unroll
    for (int kc = 0; kc < 4; ++kc)
      Aq[kc] = ldfrag(qr + kc * 32 + hf * 8, qr + kc * 32 + 16 + hf * 8);
#pragma unroll
    for (int r = 0; r < 8; ++r) {
      m_run[r] = -1e30f;
      l_run[r] = 0.0f;
    }
  }
  v8f O0 = (v8f){0, 0, 0, 0, 0, 0, 0, 0};
  v8f O1 = (v8f){0, 0, 0, 0, 0, 0, 0, 0};

  const u16_t* kb = kbf + ((size_t)b * NKVH + kh) * (size_t)KVL * HDIM;
  const u16_t* vb = vbf + ((size_t)b * NKVH + kh) * (size_t)KVL * HDIM;

  const int row = tid >> 2, d0 = (tid & 3) * 32;  // chunk loader mapping
  const unsigned ldsK = (unsigned)(unsigned long long)(const void*)&Ks[row][d0];

  for (int c = 0; c < KVL / 64; ++c) {
    // ---- K: async bf16 global -> LDS (64B per thread) ----
    async_b128(ldsK, kb + (size_t)(c * 64 + row) * HDIM + d0);
    // ---- V: bf16 global -> registers -> transposed LDS ----
    {
      const uint4* sv = (const uint4*)(vb + (size_t)(c * 64 + row) * HDIM + d0);
      uint4 q0 = sv[0], q1 = sv[1], q2 = sv[2], q3 = sv[3];
      u16_t tmp[32];
      *(uint4*)&tmp[0] = q0;
      *(uint4*)&tmp[8] = q1;
      *(uint4*)&tmp[16] = q2;
      *(uint4*)&tmp[24] = q3;
#pragma unroll
      for (int i = 0; i < 32; ++i) Vt[d0 + i][row] = tmp[i];
    }
    if (c + 1 < KVL / 64) {  // prefetch next chunk (global_prefetch_b8)
      __builtin_prefetch(kb + (size_t)(c * 64 + 64 + row) * HDIM + d0, 0, 1);
      __builtin_prefetch(vb + (size_t)(c * 64 + 64 + row) * HDIM + d0, 0, 1);
    }
    wait_async0();
    __syncthreads();

    // ---- score waves: QK^T + online softmax over 64 columns ----
    if (dt == 0) {
      float st[4][8];
#pragma unroll
      for (int j = 0; j < 4; ++j) {
        v8f acc = (v8f){0, 0, 0, 0, 0, 0, 0, 0};
#pragma unroll
        for (int kc = 0; kc < 4; ++kc) {
          const u16_t* kr = &Ks[j * 16 + l15][kc * 32 + hf * 8];
          v16bf bf_ = ldfrag(kr, kr + 16);
          acc = wmma_bf16(Aq[kc], bf_, acc);
        }
#pragma unroll
        for (int r = 0; r < 8; ++r) {
          const int qrow = qt * 16 + r + 8 * hf;
          const float mv =
              mask[((size_t)b * NS + qrow) * KVL + (size_t)c * 64 + j * 16 + l15];
          st[j][r] = acc[r] * ATT_SCALE + mv;
        }
      }
#pragma unroll
      for (int r = 0; r < 8; ++r) {
        float rm = fmaxf(fmaxf(st[0][r], st[1][r]), fmaxf(st[2][r], st[3][r]));
#pragma unroll
        for (int off = 1; off < 16; off <<= 1)
          rm = fmaxf(rm, __shfl_xor(rm, off, 32));
        const float mn = fmaxf(m_run[r], rm);
        const float corr = __expf(m_run[r] - mn);
        float pj[4];
        float rs = 0.0f;
#pragma unroll
        for (int j = 0; j < 4; ++j) {
          pj[j] = __expf(st[j][r] - mn);
          rs += pj[j];
        }
#pragma unroll
        for (int off = 1; off < 16; off <<= 1) rs += __shfl_xor(rs, off, 32);
        l_run[r] = l_run[r] * corr + rs;
        m_run[r] = mn;
        const int qrow = qt * 16 + r + 8 * hf;
#pragma unroll
        for (int j = 0; j < 4; ++j) Ps[qrow][j * 16 + l15] = f2bf(pj[j]);
        if (l15 == 0) cfac[qrow] = corr;
      }
    }
    __syncthreads();

    // ---- all waves: rescale O, then P*V (two K-halves of 32) ----
    {
      const u16_t* pr = &Ps[qt * 16 + l15][0];
      v16bf Pf0 = ldfrag(pr + hf * 8, pr + 16 + hf * 8);
      v16bf Pf1 = ldfrag(pr + 32 + hf * 8, pr + 48 + hf * 8);
#pragma unroll
      for (int r = 0; r < 8; ++r) {
        const float cf = cfac[qt * 16 + r + 8 * hf];
        O0[r] *= cf;
        O1[r] *= cf;
      }
      const u16_t* v0 = &Vt[dt * 32 + l15][0];
      O0 = wmma_bf16(Pf0, ldfrag(v0 + hf * 8, v0 + 16 + hf * 8), O0);
      O0 = wmma_bf16(Pf1, ldfrag(v0 + 32 + hf * 8, v0 + 48 + hf * 8), O0);
      const u16_t* v1 = &Vt[dt * 32 + 16 + l15][0];
      O1 = wmma_bf16(Pf0, ldfrag(v1 + hf * 8, v1 + 16 + hf * 8), O1);
      O1 = wmma_bf16(Pf1, ldfrag(v1 + 32 + hf * 8, v1 + 48 + hf * 8), O1);
    }
    __syncthreads();
  }

  if (dt == 0 && l15 == 0) {
#pragma unroll
    for (int r = 0; r < 8; ++r) lsum[qt * 16 + r + 8 * hf] = l_run[r];
  }
  __syncthreads();

#pragma unroll
  for (int r = 0; r < 8; ++r) {
    const int srow = qt * 16 + r + 8 * hf;
    const float inv = 1.0f / lsum[srow];
    const size_t base = (size_t)(b * NS + srow) * (NHEAD * HDIM) + hh * HDIM;
    abf[base + dt * 32 + l15] = f2bf(O0[r] * inv);
    abf[base + dt * 32 + 16 + l15] = f2bf(O1[r] * inv);
  }
}

// =============================================================================
extern "C" void kernel_launch(void* const* d_in, const int* in_sizes, int n_in,
                              void* d_out, int out_size, void* d_ws,
                              size_t ws_size, hipStream_t stream) {
  (void)in_sizes; (void)n_in; (void)out_size; (void)ws_size;
  const float* hidden = (const float*)d_in[0];
  const float* cosb   = (const float*)d_in[1];
  const float* sinb   = (const float*)d_in[2];
  const float* kcache = (const float*)d_in[3];
  const float* vcache = (const float*)d_in[4];
  const float* amask  = (const float*)d_in[5];
  const int*   rp     = (const int*)d_in[6];
  const float* q_w = (const float*)d_in[7];
  const float* q_b = (const float*)d_in[8];
  const float* k_w = (const float*)d_in[9];
  const float* k_b = (const float*)d_in[10];
  const float* v_w = (const float*)d_in[11];
  const float* v_b = (const float*)d_in[12];
  const float* o_w = (const float*)d_in[13];
  float* out = (float*)d_out;

  char* wsb = (char*)d_ws;
  size_t off = 0;
  auto carve = [&](size_t bytes) {
    char* p = wsb + off;
    off = (off + bytes + 255) & ~(size_t)255;
    return p;
  };
  const int MROWS = NB * NS;  // 256
  u16_t* hbf  = (u16_t*)carve((size_t)MROWS * NHID * 2);
  u16_t* qwbf = (u16_t*)carve((size_t)NHID * NHID * 2);
  u16_t* kwbf = (u16_t*)carve((size_t)NHID * (NKVH * HDIM) * 2);
  u16_t* vwbf = (u16_t*)carve((size_t)NHID * (NKVH * HDIM) * 2);
  u16_t* owbf = (u16_t*)carve((size_t)NHID * NHID * 2);
  float* qf   = (float*)carve((size_t)MROWS * NHID * 4);
  float* kf   = (float*)carve((size_t)MROWS * (NKVH * HDIM) * 4);
  float* vf   = (float*)carve((size_t)MROWS * (NKVH * HDIM) * 4);
  u16_t* qpk  = (u16_t*)carve((size_t)NB * NHEAD * NS * HDIM * 2);
  u16_t* knbf = (u16_t*)carve((size_t)NB * NKVH * NS * HDIM * 2);
  u16_t* vnbf = (u16_t*)carve((size_t)NB * NKVH * NS * HDIM * 2);
  u16_t* abf  = (u16_t*)carve((size_t)MROWS * NHID * 2);
  u16_t* kbf  = (u16_t*)carve((size_t)NB * NKVH * KVL * HDIM * 2);
  u16_t* vbf  = (u16_t*)carve((size_t)NB * NKVH * KVL * HDIM * 2);

  auto cvt = [&](const float* s, u16_t* d, int n) {
    int n8 = n / 8;
    int blocks = (n8 + 255) / 256;
    if (blocks > 4096) blocks = 4096;
    cvt_kernel<<<blocks, 256, 0, stream>>>(s, d, n8);
  };
  cvt(hidden, hbf, MROWS * NHID);
  cvt(q_w, qwbf, NHID * NHID);
  cvt(k_w, kwbf, NHID * NKVH * HDIM);
  cvt(v_w, vwbf, NHID * NKVH * HDIM);
  cvt(o_w, owbf, NHID * NHID);

  // QKV projections (bf16 WMMA, fp32 out + bias)
  gemm_bf16<<<dim3(MROWS / 64, NHID / 64), 128, 0, stream>>>(
      hbf, qwbf, q_b, qf, MROWS, NHID, NHID);
  gemm_bf16<<<dim3(MROWS / 64, (NKVH * HDIM) / 64), 128, 0, stream>>>(
      hbf, kwbf, k_b, kf, MROWS, NKVH * HDIM, NHID);
  gemm_bf16<<<dim3(MROWS / 64, (NKVH * HDIM) / 64), 128, 0, stream>>>(
      hbf, vwbf, v_b, vf, MROWS, NKVH * HDIM, NHID);

  // RoPE + pack to per-head bf16
  {
    int tq = NB * NS * NHEAD * 64;
    rope_pack<<<(tq + 255) / 256, 256, 0, stream>>>(qf, cosb, sinb, qpk, NHEAD, 0, tq);
    int tk = NB * NS * NKVH * 64;
    rope_pack<<<(tk + 255) / 256, 256, 0, stream>>>(kf, cosb, sinb, knbf, NKVH, 0, tk);
    rope_pack<<<(tk + 255) / 256, 256, 0, stream>>>(vf, cosb, sinb, vnbf, NKVH, 1, tk);
  }

  // K/V cache -> bf16 with fresh-row splice (one pass; stays hot in 192MB L2)
  {
    int total8 = NB * NKVH * KVL * HDIM / 8;
    cvt_kv<<<(total8 + 255) / 256, 256, 0, stream>>>(kcache, knbf, kbf, rp, total8);
    cvt_kv<<<(total8 + 255) / 256, 256, 0, stream>>>(vcache, vnbf, vbf, rp, total8);
  }

  // Flash attention: one workgroup per (batch, head)
  attn_kernel<<<NB * NHEAD, 256, 0, stream>>>(qpk, kbf, vbf, amask, abf);

  // Output projection -> fp32 result
  gemm_bf16<<<dim3(MROWS / 64, NHID / 64), 128, 0, stream>>>(
      abf, owbf, nullptr, out, MROWS, NHID, NHID);
}
